// Gtu_64510408786205
// MI455X (gfx1250) — matrix operand order repaired and checked by reading, not compiled, not run on hardware
//
#include <hip/hip_runtime.h>

// ---------------------------------------------------------------------------
// MI455X (gfx1250) GTU kernel pipeline: bf16 WMMA GEMMs + TDM-staged
// per-channel causal Toeplitz convolution.
// ---------------------------------------------------------------------------

typedef unsigned int u32;
typedef __attribute__((ext_vector_type(16))) __bf16 v16bf;
typedef __attribute__((ext_vector_type(8)))  __bf16 v8bf;
typedef __attribute__((ext_vector_type(4)))  __bf16 v4bf;
typedef __attribute__((ext_vector_type(8)))  float  v8f;
typedef __attribute__((ext_vector_type(4)))  unsigned int uint32x4;
typedef __attribute__((ext_vector_type(8)))  int int32x8;
typedef __attribute__((ext_vector_type(4)))  int int32x4;

#define EMBED   1024
#define D1      1024
#define SEQ     4096
#define BATCH   4
#define ROWS    (BATCH*SEQ)      // 16384
#define FEAT    32
#define TPAD    32               // front zero pad of t (causality)
#define EXTROWS 4144             // 32 + 4096 + 16 pad rows for t and v

__device__ __forceinline__ float silu(float x) {
    return x * (1.0f / (1.0f + __expf(-x)));
}

// ---------------------------------------------------------------------------
// TDM: 1-D contiguous global->LDS DMA via Tensor Data Mover (D# built inline).
// data_size = 4 bytes (code 2); tile_dim0/tensor_dim0 in dword units.
// 6-arg builtin on this toolchain:
//   (uint32x4 g0, int32x8 g1, int32x4 g2, int32x4 g3, int32x8 pad, i32 cpol)
// ---------------------------------------------------------------------------
__device__ __forceinline__ void tdm_load_1d(u32 lds_addr, const void* gptr, u32 ndwords) {
    unsigned long long ga = (unsigned long long)(uintptr_t)gptr;
    uint32x4 g0;
    g0.x = 1u;                                           // count=1, user mode
    g0.y = lds_addr;                                     // lds_addr [63:32]
    g0.z = (u32)(ga & 0xFFFFFFFFu);                      // global_addr[31:0]
    g0.w = (u32)((ga >> 32) & 0x01FFFFFFu) | (2u << 30); // global_addr[56:32], type=2
    int32x8 g1;
    g1[0] = (int)(2u << 16);                             // data_size=4B, wg_mask=0
    g1[1] = (int)((ndwords & 0xFFFFu) << 16);            // tensor_dim0 lo16
    g1[2] = (int)(((ndwords >> 16) & 0xFFFFu) | (1u << 16)); // td0 hi16 | tensor_dim1=1
    g1[3] = (int)((ndwords & 0xFFFFu) << 16);            // tile_dim0
    g1[4] = 0;                                           // tile_dim1=0 (1-D), tile_dim2=0
    g1[5] = (int)ndwords;                                // tensor_dim0_stride lo32
    g1[6] = 0;
    g1[7] = 0;
    int32x4 z4 = {0, 0, 0, 0};
    int32x8 z8 = {0, 0, 0, 0, 0, 0, 0, 0};
    __builtin_amdgcn_tensor_load_to_lds(g0, g1, z4, z4, z8, 0);
}

// ---------------------------------------------------------------------------
// 1) RPE MLP hidden features: one wave per position i, lane = feature.
// ---------------------------------------------------------------------------
__global__ void rpe_h_kernel(const float* __restrict__ w_in,
                             const float* __restrict__ hid_w,
                             const float* __restrict__ ln_g,
                             const float* __restrict__ ln_b,
                             float* __restrict__ H) {
    const int wave = threadIdx.x >> 5;
    const int lane = threadIdx.x & 31;
    const int i = blockIdx.x * 8 + wave;           // 512 blocks * 8 waves = 4096
    float h = (float)i * w_in[lane];
    h = silu(h);
    for (int l = 0; l < 3; ++l) {
        float mu = h;
        for (int o = 16; o; o >>= 1) mu += __shfl_xor(mu, o);
        mu *= (1.0f / 32.0f);
        float d = h - mu;
        float var = d * d;
        for (int o = 16; o; o >>= 1) var += __shfl_xor(var, o);
        var *= (1.0f / 32.0f);
        float hn = d * rsqrtf(var + 1e-5f) * ln_g[l * 32 + lane] + ln_b[l * 32 + lane];
        float acc = 0.0f;
        #pragma unroll 8
        for (int f = 0; f < 32; ++f) {
            float hf = __shfl(hn, f);
            acc += hf * hid_w[(l * 32 + f) * 32 + lane];
        }
        h = silu(acc);
    }
    H[i * 32 + lane] = h;
}

// ---------------------------------------------------------------------------
// 2) Toeplitz coefficients (channel-major, decayed, zero-padded) in bf16.
// ---------------------------------------------------------------------------
__global__ void rpe_t_kernel(const float* __restrict__ H,
                             const float* __restrict__ out_w,
                             const float* __restrict__ dg,
                             __bf16* __restrict__ TEXT) {
    const int c = blockIdx.x * blockDim.x + threadIdx.x;   // 0..1023
    const float gamma = 0.99f + 0.01f * (1.0f / (1.0f + __expf(-dg[c])));
    const float lg = __logf(gamma);
    float wcol[32];
    #pragma unroll
    for (int f = 0; f < 32; ++f) wcol[f] = out_w[f * D1 + c];
    for (int m = 0; m < EXTROWS; ++m) {
        float val = 0.0f;
        const int i = m - TPAD;
        if (i >= 0 && i < SEQ) {
            float acc = 0.0f;
            #pragma unroll 8
            for (int f = 0; f < 32; ++f) acc += H[i * 32 + f] * wcol[f];
            val = acc * __expf((float)i * lg);
        }
        TEXT[(size_t)c * EXTROWS + m] = (__bf16)val;
    }
}

// ---------------------------------------------------------------------------
// 3/6) WMMA bf16 GEMM, 128x128 tile per 256-thread block (8 waves, 2x4 grid,
//      each wave 64x32 = 4x2 fragments). K-step 32.
//      As row-major [128][40] (padded), Bs column-major [128][40] so both
//      fragments are contiguous 16B/32B LDS vector loads.
// MODE 0: A=f32 (x), B=f32 (W_uv); epilogue silu, split cols into U / VRM bf16.
// MODE 1: A=bf16 (g), B=f32 (W_o);  epilogue plain f32 store.
// ---------------------------------------------------------------------------
template <int MODE>
__global__ void wmma_gemm_kernel(const void* __restrict__ Ap,
                                 const float* __restrict__ B,
                                 void* __restrict__ C0, void* __restrict__ C1,
                                 int M, int N, int K) {
    __shared__ __align__(32) __bf16 As[128][40];   // [m][k], 80B rows
    __shared__ __align__(32) __bf16 Bs[128][40];   // [n][k], 80B rows

    const int tid  = threadIdx.x;
    const int lane = tid & 31;
    const int wave = tid >> 5;
    const int wm = wave >> 2;          // 0..1
    const int wn = wave & 3;           // 0..3
    const int ln16 = lane & 15;
    const int hi = lane >> 4;

    const int n0 = blockIdx.x * 128;
    const int m0 = blockIdx.y * 128;

    v8f acc[4][2];
    #pragma unroll
    for (int fm = 0; fm < 4; ++fm)
        #pragma unroll
        for (int fn = 0; fn < 2; ++fn) acc[fm][fn] = (v8f)0.0f;

    for (int k0 = 0; k0 < K; k0 += 32) {
        // stage A (128 rows x 32 k) as bf16, float4/v4bf vectorized
        for (int e4 = tid * 4; e4 < 128 * 32; e4 += 1024) {
            const int r = e4 >> 5, kk = e4 & 31;
            v4bf o;
            if constexpr (MODE == 0) {
                const float4 f =
                    *(const float4*)&((const float*)Ap)[(size_t)(m0 + r) * K + k0 + kk];
                o[0] = (__bf16)f.x; o[1] = (__bf16)f.y;
                o[2] = (__bf16)f.z; o[3] = (__bf16)f.w;
            } else {
                o = *(const v4bf*)&((const __bf16*)Ap)[(size_t)(m0 + r) * K + k0 + kk];
            }
            *(v4bf*)&As[r][kk] = o;
        }
        // stage B (32 k x 128 n) as bf16 into column-major Bs[n][k]
        for (int e4 = tid * 4; e4 < 32 * 128; e4 += 1024) {
            const int r = e4 >> 7, nn = e4 & 127;
            const float4 f = *(const float4*)&B[(size_t)(k0 + r) * N + n0 + nn];
            Bs[nn + 0][r] = (__bf16)f.x;
            Bs[nn + 1][r] = (__bf16)f.y;
            Bs[nn + 2][r] = (__bf16)f.z;
            Bs[nn + 3][r] = (__bf16)f.w;
        }
        __syncthreads();

        v16bf afr[4], bfr[2];
        #pragma unroll
        for (int fm = 0; fm < 4; ++fm) {
            const int row = wm * 64 + fm * 16 + ln16;
            const v8bf c0 = *(const v8bf*)&As[row][hi * 16];       // K = hi*16 .. +7
            const v8bf c1 = *(const v8bf*)&As[row][hi * 16 + 8];   // K = hi*16+8 .. +15
            afr[fm] = __builtin_shufflevector(c0, c1, 0, 1, 2, 3, 4, 5, 6, 7,
                                              8, 9, 10, 11, 12, 13, 14, 15);
        }
        #pragma unroll
        for (int fn = 0; fn < 2; ++fn) {
            const int col = wn * 32 + fn * 16 + ln16;
            bfr[fn] = *(const v16bf*)&Bs[col][hi * 16];            // K = hi*16 .. +15
        }
        #pragma unroll
        for (int fm = 0; fm < 4; ++fm)
            #pragma unroll
            for (int fn = 0; fn < 2; ++fn)
                acc[fm][fn] = __builtin_amdgcn_wmma_f32_16x16x32_bf16(
                    false, afr[fm], false, bfr[fn], (short)0, acc[fm][fn],
                    false, false);
        __syncthreads();
    }

    // epilogue
    #pragma unroll
    for (int fm = 0; fm < 4; ++fm) {
        #pragma unroll
        for (int fn = 0; fn < 2; ++fn) {
            #pragma unroll
            for (int v = 0; v < 8; ++v) {
                const int row = m0 + wm * 64 + fm * 16 + v + hi * 8;
                const int col = n0 + wn * 32 + fn * 16 + ln16;
                const float val = acc[fm][fn][v];
                if constexpr (MODE == 0) {
                    const float s = silu(val);
                    const int half = N >> 1;   // 1024
                    if (col < half)
                        ((__bf16*)C0)[(size_t)row * half + col] = (__bf16)s;
                    else
                        ((__bf16*)C1)[(size_t)row * half + (col - half)] = (__bf16)s;
                } else {
                    ((float*)C0)[(size_t)row * N + col] = val;
                }
            }
        }
    }
}

// ---------------------------------------------------------------------------
// 4) Transpose v to channel-major padded layout VEXT[c][b][m], m = i + 16.
// ---------------------------------------------------------------------------
__global__ void transpose_v_kernel(const __bf16* __restrict__ VRM,
                                   __bf16* __restrict__ VEXT) {
    __shared__ __bf16 tile[64][65];
    const int ct = blockIdx.x;   // channel tile (16)
    const int rt = blockIdx.y;   // row tile (65)
    const int b  = blockIdx.z;   // batch (4)
    for (int e = threadIdx.x; e < 64 * 64; e += 256) {
        const int r = e >> 6, cc = e & 63;
        const int m = rt * 64 + r;
        const int i = m - 16;
        float v = 0.0f;
        if (i >= 0 && i < SEQ && m < EXTROWS)
            v = (float)VRM[((size_t)b * SEQ + i) * D1 + ct * 64 + cc];
        tile[r][cc] = (__bf16)v;
    }
    __syncthreads();
    for (int e = threadIdx.x; e < 64 * 64; e += 256) {
        const int cr = e >> 6, mm = e & 63;
        const int m = rt * 64 + mm;
        if (m < EXTROWS)
            VEXT[(((size_t)(ct * 64 + cr)) * BATCH + b) * EXTROWS + m] = tile[mm][cr];
    }
}

// ---------------------------------------------------------------------------
// 5) Causal Toeplitz conv per channel with WMMA + TDM-staged LDS, fused u-gate.
//    Wave handles 16-row output blocks I; per 32-wide j-chunk s:
//      A[p,k] = tS[16I + 48 - 32s + p - k]   (Toeplitz gather, branch-free)
//      B[k,n] = vS[(n&3)][32s + k]  (cols 4..15 compute dup results, never stored)
//    Front zero pad of tS (32 rows) makes non-causal lags contribute zero.
// ---------------------------------------------------------------------------
__global__ void toeplitz_kernel(const __bf16* __restrict__ TEXT,
                                const __bf16* __restrict__ VEXT,
                                __bf16* __restrict__ U) {
    __shared__ __align__(32) __bf16 tS[EXTROWS];
    __shared__ __align__(32) __bf16 vS[BATCH * EXTROWS];

    const int c = blockIdx.x;
    if (threadIdx.x == 0) {
        tdm_load_1d((u32)(uintptr_t)&tS[0],
                    TEXT + (size_t)c * EXTROWS, (EXTROWS * 2) / 4);
        tdm_load_1d((u32)(uintptr_t)&vS[0],
                    VEXT + (size_t)c * BATCH * EXTROWS, (BATCH * EXTROWS * 2) / 4);
        __builtin_amdgcn_s_wait_tensorcnt(0);
    }
    __syncthreads();

    const int lane = threadIdx.x & 31;
    const int wave = threadIdx.x >> 5;
    const int ln16 = lane & 15;
    const int hi = lane >> 4;
    const int bsel = ln16 & 3;     // batch column; cols 4..15 are dup garbage

    for (int t = 0; t < 32; ++t) {
        const int I = t * 8 + wave;                 // 0..255
        v8f acc = (v8f)0.0f;
        const int smax = (16 * I + 31) >> 5;
        for (int s = 0; s <= smax; ++s) {
            const int base = 16 * I + 48 - 32 * s;
            // A: Toeplitz gather (reversed in k -> scalar u16 loads, branch-free)
            v16bf a;
            #pragma unroll
            for (int e = 0; e < 16; ++e) {
                const int kk = e + hi * 16;
                a[e] = tS[base + ln16 - kk];
            }
            // B: contiguous 32B aligned vector load
            const v16bf b = *(const v16bf*)&vS[bsel * EXTROWS + 32 * s + hi * 16];
            acc = __builtin_amdgcn_wmma_f32_16x16x32_bf16(
                false, a, false, b, (short)0, acc, false, false);
        }
        // gate with u and write g = u * o in place (bf16)
        if (ln16 < BATCH) {
            #pragma unroll
            for (int v = 0; v < 8; ++v) {
                const int i = 16 * I + v + hi * 8;
                const size_t idx = ((size_t)ln16 * SEQ + i) * D1 + c;
                const float uf = (float)U[idx];
                U[idx] = (__bf16)(uf * acc[v]);
            }
        }
    }
}

// ---------------------------------------------------------------------------
// Workspace layout
// ---------------------------------------------------------------------------
static constexpr size_t OFF_H  = 0;                                    // 4096*32 f32
static constexpr size_t OFF_T  = OFF_H + (size_t)SEQ * FEAT * 4;       // 1024*4144 bf16
static constexpr size_t OFF_VX = OFF_T + (size_t)D1 * EXTROWS * 2;     // 1024*4*4144 bf16
static constexpr size_t OFF_U  = OFF_VX + (size_t)D1 * BATCH * EXTROWS * 2; // 16384*1024 bf16
static constexpr size_t OFF_VR = OFF_U + (size_t)ROWS * D1 * 2;        // 16384*1024 bf16

extern "C" void kernel_launch(void* const* d_in, const int* in_sizes, int n_in,
                              void* d_out, int out_size, void* d_ws, size_t ws_size,
                              hipStream_t stream) {
    const float* x        = (const float*)d_in[0];
    const float* W_uv     = (const float*)d_in[1];
    const float* W_o      = (const float*)d_in[2];
    const float* rpe_in_w = (const float*)d_in[3];
    const float* rpe_hid  = (const float*)d_in[4];
    const float* rpe_ln_g = (const float*)d_in[5];
    const float* rpe_ln_b = (const float*)d_in[6];
    const float* rpe_outw = (const float*)d_in[7];
    const float* decay_g  = (const float*)d_in[8];
    float* out = (float*)d_out;

    char* ws = (char*)d_ws;
    float*  H    = (float*)(ws + OFF_H);
    __bf16* TEXT = (__bf16*)(ws + OFF_T);
    __bf16* VEXT = (__bf16*)(ws + OFF_VX);
    __bf16* U    = (__bf16*)(ws + OFF_U);
    __bf16* VRM  = (__bf16*)(ws + OFF_VR);

    // 1) RPE hidden features
    rpe_h_kernel<<<dim3(SEQ / 8), dim3(256), 0, stream>>>(
        rpe_in_w, rpe_hid, rpe_ln_g, rpe_ln_b, H);

    // 2) decayed Toeplitz coefficients (channel-major, padded)
    rpe_t_kernel<<<dim3(D1 / 256), dim3(256), 0, stream>>>(
        H, rpe_outw, decay_g, TEXT);

    // 3) uv = silu(x @ W_uv), split into u / v (bf16)
    wmma_gemm_kernel<0><<<dim3(2 * D1 / 128, ROWS / 128), dim3(256), 0, stream>>>(
        (const void*)x, W_uv, (void*)U, (void*)VRM, ROWS, 2 * D1, EMBED);

    // 4) v -> channel-major padded
    transpose_v_kernel<<<dim3(D1 / 64, (EXTROWS + 63) / 64, BATCH),
                         dim3(256), 0, stream>>>(VRM, VEXT);

    // 5) o = causal Toeplitz conv; g = u * o (in place in U)
    toeplitz_kernel<<<dim3(D1), dim3(256), 0, stream>>>(TEXT, VEXT, U);

    // 6) out = g @ W_o (f32)
    wmma_gemm_kernel<1><<<dim3(EMBED / 128, ROWS / 128), dim3(256), 0, stream>>>(
        (const void*)U, W_o, (void*)out, nullptr, ROWS, EMBED, D1);
}
